// AttentionDecoder_38895223832781
// MI455X (gfx1250) — matrix operand order repaired and checked by reading, not compiled, not run on hardware
//
#include <hip/hip_runtime.h>

// ---------------------------------------------------------------------------
// MI455X (gfx1250, wave32) attention-decoder step.
// Dominant work: enc(B*S x 512) @ We^T  (137 GFLOP) fused with tanh/v_w
// score reduction, done with v_wmma_f32_16x16x32_bf16 (fp32 accumulate).
// Memory floor: enc read twice (energy + context) ~ 1 GB / 23.3 TB/s ~ 45us.
// 64x128 block tile, 8 waves: each wave = 1 m-tile x 4 n-tiles (4 WMMAs per
// K-step reusing one A fragment). K-loop double-buffered through LDS with
// register prefetch; one barrier per K-step; native v_cvt_pk_bf16_f32.
// ---------------------------------------------------------------------------

typedef __attribute__((ext_vector_type(8)))  float  v8f;
typedef __attribute__((ext_vector_type(16))) __bf16 v16bf;

union FragBF { unsigned int u[8]; v16bf v; };

// One packed RTNE convert per staged dword.
__device__ __forceinline__ unsigned int pack_bf16x2(float lo, float hi) {
#if __has_builtin(__builtin_amdgcn_cvt_pk_bf16_f32)
  typedef __attribute__((ext_vector_type(2))) __bf16 v2bf;
  v2bf r = __builtin_amdgcn_cvt_pk_bf16_f32(lo, hi);
  unsigned int u;
  __builtin_memcpy(&u, &r, 4);
  return u;
#else
  union { __bf16 h[2]; unsigned int u; } cv;
  cv.h[0] = (__bf16)lo;
  cv.h[1] = (__bf16)hi;
  return cv.u;
#endif
}

__device__ __forceinline__ float sigmf(float x) { return 1.0f / (1.0f + __expf(-x)); }

// ---------------------------------------------------------------------------
// Generic 64x128-tile bf16 WMMA GEMM:  out = A(MxK,fp32) * B(NxK,fp32)^T
// 256 threads = 8 waves; wave w: m-tile (w&3), n-tiles {4*(w>>2)+t, t=0..3}.
// MODE 0: out[m*ldo+n] = acc + bias[n]
// MODE 1: score epilogue: atomicAdd(out[z*strideOz+m],
//                                   sum_n tanh(acc + qb2[z][n]) * vw[n])
// ---------------------------------------------------------------------------
template <int MODE>
__global__ __launch_bounds__(256) void gemm_bf16_tile(
    const float* __restrict__ A, int lda, long long strideAz,
    const float* __restrict__ B, int ldb,
    const float* __restrict__ bias, long long strideBiasZ,
    const float* __restrict__ vw,
    float* __restrict__ out, long long ldo, long long strideOz,
    int K) {
  // Double-buffered staging: [buf][row][kpair], row stride 20 dwords (pad).
  __shared__ __align__(16) unsigned int ldsA[2][64 * 20];
  __shared__ __align__(16) unsigned int ldsB[2][128 * 20];

  const int tid  = threadIdx.x;
  const int wave = tid >> 5;
  const int lane = tid & 31;
  const int half = lane >> 4;     // 0: lanes 0-15, 1: lanes 16-31
  const int l16  = lane & 15;
  const int wm   = wave & 3;      // m-tile within 64-row block
  const int wq   = wave >> 2;     // n-quad selector (0/1): n-tiles wq*4+t
  const int m0   = blockIdx.x * 64;
  const int n0   = blockIdx.y * 128;
  const int z    = blockIdx.z;

  const float* Az = A + (long long)z * strideAz;

  // Per-thread staging coordinates (rows x 16 kpairs split over 256 thr).
  const int srow = tid >> 4;      // 0..15 base row, +16 per i
  const int skp  = tid & 15;      // kpair 0..15

  v8f acc[4];
#pragma unroll
  for (int t = 0; t < 4; ++t) acc[t] = v8f{};

  const int nk = K >> 5;          // K-steps of 32
  float2 ra[4], rb[8];

  // Prologue: stage chunk 0.
#pragma unroll
  for (int i = 0; i < 4; ++i)
    ra[i] = *(const float2*)(Az + (size_t)(m0 + srow + i * 16) * lda + 2 * skp);
#pragma unroll
  for (int i = 0; i < 8; ++i)
    rb[i] = *(const float2*)(B + (size_t)(n0 + srow + i * 16) * ldb + 2 * skp);
#pragma unroll
  for (int i = 0; i < 4; ++i)
    ldsA[0][(srow + i * 16) * 20 + skp] = pack_bf16x2(ra[i].x, ra[i].y);
#pragma unroll
  for (int i = 0; i < 8; ++i)
    ldsB[0][(srow + i * 16) * 20 + skp] = pack_bf16x2(rb[i].x, rb[i].y);
  __syncthreads();

  for (int ik = 0; ik < nk; ++ik) {
    const int p = ik & 1;
    const int q = p ^ 1;

    // Prefetch next chunk into registers (latency hidden by the WMMAs below).
    if (ik + 1 < nk) {
      const int kk = (ik + 1) << 5;
#pragma unroll
      for (int i = 0; i < 4; ++i)
        ra[i] = *(const float2*)(Az + (size_t)(m0 + srow + i * 16) * lda + kk + 2 * skp);
#pragma unroll
      for (int i = 0; i < 8; ++i)
        rb[i] = *(const float2*)(B + (size_t)(n0 + srow + i * 16) * ldb + kk + 2 * skp);
    }

    // Compute current chunk.
    // A fragment: 16x32 bf16. kpair(v,half): v<4 -> v+4*half ; v>=4 -> 8+(v-4)+4*half
    FragBF fa;
    const int arow = wm * 16 + l16;
    *(uint4*)&fa.u[0] = *(const uint4*)&ldsA[p][arow * 20 + half * 4];
    *(uint4*)&fa.u[4] = *(const uint4*)&ldsA[p][arow * 20 + 8 + half * 4];

    // B fragments: 32x16 bf16. kpair(v,half): v<4 -> v+4*half ; v>=4 -> 8+(v-4)+4*half
#pragma unroll
    for (int t = 0; t < 4; ++t) {
      FragBF fb;
      const int bn = (wq * 4 + t) * 16 + l16;
      *(uint4*)&fb.u[0] = *(const uint4*)&ldsB[p][bn * 20 + half * 4];
      *(uint4*)&fb.u[4] = *(const uint4*)&ldsB[p][bn * 20 + 8 + half * 4];
      acc[t] = __builtin_amdgcn_wmma_f32_16x16x32_bf16(false, fa.v, false, fb.v,
                                                       (short)0, acc[t], false, false);
    }

    // Convert + store next chunk into the alternate buffer, then one barrier.
    // (read(ik-1) of buf q completed before barrier(ik-1), so this is safe.)
    if (ik + 1 < nk) {
#pragma unroll
      for (int i = 0; i < 4; ++i)
        ldsA[q][(srow + i * 16) * 20 + skp] = pack_bf16x2(ra[i].x, ra[i].y);
#pragma unroll
      for (int i = 0; i < 8; ++i)
        ldsB[q][(srow + i * 16) * 20 + skp] = pack_bf16x2(rb[i].x, rb[i].y);
      __syncthreads();
    }
  }

  // Epilogues. C/D layout: VGPR r -> row (r + 8*half), col (lane&15).
  if (MODE == 0) {
#pragma unroll
    for (int t = 0; t < 4; ++t) {
      const int col = n0 + (wq * 4 + t) * 16 + l16;
      const float bv = bias[col];
#pragma unroll
      for (int r = 0; r < 8; ++r) {
        const int row = m0 + wm * 16 + half * 8 + r;
        out[(size_t)row * ldo + col] = acc[t][r] + bv;
      }
    }
  } else {
    float p[8] = {0, 0, 0, 0, 0, 0, 0, 0};
#pragma unroll
    for (int t = 0; t < 4; ++t) {
      const int col = n0 + (wq * 4 + t) * 16 + l16;
      const float q = bias[(long long)z * strideBiasZ + col];
      const float w = vw[col];
#pragma unroll
      for (int r = 0; r < 8; ++r) p[r] += tanhf(acc[t][r] + q) * w;
    }
#pragma unroll
    for (int r = 0; r < 8; ++r) {
#pragma unroll
      for (int mask = 1; mask < 16; mask <<= 1)
        p[r] += __shfl_xor(p[r], mask, 32);
    }
    if (l16 == 0) {
      float* orow = out + (long long)z * strideOz + m0 + wm * 16 + half * 8;
#pragma unroll
      for (int r = 0; r < 8; ++r) atomicAdd(&orow[r], p[r]);
    }
  }
}

// ---------------------------------------------------------------------------
__global__ void zero_kernel(float* __restrict__ p, int n) {
  const int i = blockIdx.x * blockDim.x + threadIdx.x;
  if (i < n) p[i] = 0.0f;
}

// Wcat[n][0:1024]=W_ih[n], Wcat[n][1024:1536]=W_hh[n]; bsum = b_ih + b_hh
__global__ void prep_cat_kernel(const float* __restrict__ W_ih,
                                const float* __restrict__ W_hh,
                                const float* __restrict__ b_ih,
                                const float* __restrict__ b_hh,
                                float* __restrict__ Wcat,
                                float* __restrict__ bsum) {
  const int gid = blockIdx.x * blockDim.x + threadIdx.x;
  const int total = 2048 * 1536;
  if (gid < total) {
    const int n = gid / 1536;
    const int k = gid - n * 1536;
    Wcat[gid] = (k < 1024) ? W_ih[(size_t)n * 1024 + k]
                           : W_hh[(size_t)n * 512 + (k - 1024)];
  }
  if (gid < 2048) bsum[gid] = b_ih[gid] + b_hh[gid];
}

// Row-wise softmax over S=1024 per batch element.
__global__ __launch_bounds__(256) void softmax_kernel(
    const float* __restrict__ scores, float* __restrict__ attn) {
  const int b = blockIdx.x, tid = threadIdx.x;
  __shared__ float red[256];
  const float* row = scores + (size_t)b * 1024;
  float v0 = row[tid], v1 = row[tid + 256], v2 = row[tid + 512], v3 = row[tid + 768];
  float mx = fmaxf(fmaxf(v0, v1), fmaxf(v2, v3));
  red[tid] = mx;
  __syncthreads();
  for (int s = 128; s > 0; s >>= 1) {
    if (tid < s) red[tid] = fmaxf(red[tid], red[tid + s]);
    __syncthreads();
  }
  mx = red[0];
  __syncthreads();
  const float e0 = __expf(v0 - mx), e1 = __expf(v1 - mx);
  const float e2 = __expf(v2 - mx), e3 = __expf(v3 - mx);
  red[tid] = e0 + e1 + e2 + e3;
  __syncthreads();
  for (int s = 128; s > 0; s >>= 1) {
    if (tid < s) red[tid] += red[tid + s];
    __syncthreads();
  }
  const float inv = 1.0f / red[0];
  float* o = attn + (size_t)b * 1024;
  o[tid] = e0 * inv; o[tid + 256] = e1 * inv;
  o[tid + 512] = e2 * inv; o[tid + 768] = e3 * inv;
}

// context[b][e] = sum_s attn[b][s] * enc[b][s][e]   (memory-bound pass)
__global__ __launch_bounds__(256) void context_kernel(
    const float* __restrict__ enc, const float* __restrict__ attn,
    float* __restrict__ ctx) {
  const int b = blockIdx.x, tid = threadIdx.x;
  __shared__ float w[1024];
#pragma unroll
  for (int i = 0; i < 4; ++i) w[tid + i * 256] = attn[(size_t)b * 1024 + tid + i * 256];
  __syncthreads();
  const float* base = enc + (size_t)b * 1024 * 512;
  float a0 = 0.0f, a1 = 0.0f;
  for (int s = 0; s < 1024; ++s) {
    if (s + 8 < 1024)  // emits global_prefetch_b8
      __builtin_prefetch(base + (size_t)(s + 8) * 512 + tid, 0, 0);
    const float ws = w[s];
    a0 += ws * base[(size_t)s * 512 + tid];
    a1 += ws * base[(size_t)s * 512 + tid + 256];
  }
  ctx[(size_t)b * 512 + tid] = a0;
  ctx[(size_t)b * 512 + tid + 256] = a1;
}

// Xg[b] = [ emb_table[x[b]] | ctx[b] | h0[b] ]  (matches [W_ih | W_hh] K-order)
__global__ void build_xg_kernel(const int* __restrict__ x,
                                const float* __restrict__ emb,
                                const float* __restrict__ ctx,
                                const float* __restrict__ h0,
                                float* __restrict__ Xg) {
  const int gid = blockIdx.x * blockDim.x + threadIdx.x;  // < 256*1536
  const int b = gid / 1536;
  const int k = gid - b * 1536;
  float v;
  if (k < 512)       v = emb[(size_t)x[b] * 512 + k];
  else if (k < 1024) v = ctx[(size_t)b * 512 + (k - 512)];
  else               v = h0[(size_t)b * 512 + (k - 1024)];
  Xg[gid] = v;
}

// Xf[b] = [ h1[b] | ctx[b] | emb_table[x[b]] ]  (fc_W K-order)
__global__ void build_xf_kernel(const int* __restrict__ x,
                                const float* __restrict__ emb,
                                const float* __restrict__ ctx,
                                const float* __restrict__ h1,
                                float* __restrict__ Xf) {
  const int gid = blockIdx.x * blockDim.x + threadIdx.x;
  const int b = gid / 1536;
  const int k = gid - b * 1536;
  float v;
  if (k < 512)       v = h1[(size_t)b * 512 + k];
  else if (k < 1024) v = ctx[(size_t)b * 512 + (k - 512)];
  else               v = emb[(size_t)x[b] * 512 + (k - 1024)];
  Xf[gid] = v;
}

// Pointwise LSTM cell from gates (B x 2048, [i|f|g|o]).
__global__ void lstm_kernel(const float* __restrict__ gates,
                            const float* __restrict__ c0,
                            float* __restrict__ h1, float* __restrict__ c1) {
  const int gid = blockIdx.x * blockDim.x + threadIdx.x;  // < 256*512
  const int b = gid >> 9;
  const int j = gid & 511;
  const float* g = gates + (size_t)b * 2048;
  const float ig = sigmf(g[j]);
  const float fg = sigmf(g[512 + j]);
  const float gg = tanhf(g[1024 + j]);
  const float og = sigmf(g[1536 + j]);
  const float c = fg * c0[gid] + ig * gg;
  c1[gid] = c;
  h1[gid] = og * tanhf(c);
}

// ---------------------------------------------------------------------------
extern "C" void kernel_launch(void* const* d_in, const int* in_sizes, int n_in,
                              void* d_out, int out_size, void* d_ws, size_t ws_size,
                              hipStream_t stream) {
  (void)in_sizes; (void)n_in; (void)out_size; (void)ws_size;
  const int*   x      = (const int*)d_in[0];
  const float* h0     = (const float*)d_in[1];
  const float* c0     = (const float*)d_in[2];
  const float* enc    = (const float*)d_in[3];
  const float* emb    = (const float*)d_in[4];
  const float* attn_W = (const float*)d_in[5];
  const float* attn_b = (const float*)d_in[6];
  const float* v_w    = (const float*)d_in[7];
  const float* W_ih   = (const float*)d_in[8];
  const float* W_hh   = (const float*)d_in[9];
  const float* b_ih   = (const float*)d_in[10];
  const float* b_hh   = (const float*)d_in[11];
  const float* fc_W   = (const float*)d_in[12];
  const float* fc_b   = (const float*)d_in[13];

  float* out    = (float*)d_out;
  float* logits = out;                   // 256*32000
  float* h1     = out + 8192000;         // 256*512
  float* c1     = h1 + 131072;           // 256*512
  float* attn   = c1 + 131072;           // 256*1024

  float* ws     = (float*)d_ws;
  float* qb2    = ws;                    // 256*512
  float* scores = qb2 + 131072;          // 256*1024
  float* ctx    = scores + 262144;       // 256*512
  float* Wcat   = ctx + 131072;          // 2048*1536
  float* bsum   = Wcat + 3145728;        // 2048
  float* Xg     = bsum + 2048;           // 256*1536
  float* gates  = Xg + 393216;           // 256*2048
  float* Xf     = gates + 524288;        // 256*1536

  // 0) zero scores accumulator (score epilogue uses atomicAdd).
  zero_kernel<<<1024, 256, 0, stream>>>(scores, 262144);

  // 1) concat weights for the gate GEMM.
  prep_cat_kernel<<<(2048 * 1536 + 255) / 256, 256, 0, stream>>>(
      W_ih, W_hh, b_ih, b_hh, Wcat, bsum);

  // 2) qb2 = h0 @ Wh^T + attn_b   (Wh = attn_W[:, :512], ldb=1024)
  gemm_bf16_tile<0><<<dim3(4, 4, 1), 256, 0, stream>>>(
      h0, 512, 0, attn_W, 1024, attn_b, 0, nullptr, qb2, 512, 0, 512);

  // 3) scores[b,s] = sum_d v_w[d]*tanh(enc[b]@We^T + qb2[b])  (We ldb=1024)
  gemm_bf16_tile<1><<<dim3(16, 4, 256), 256, 0, stream>>>(
      enc, 512, (long long)1024 * 512, attn_W + 512, 1024,
      qb2, 512, v_w, scores, 0, 1024, 512);

  // 4) softmax over S -> attn output.
  softmax_kernel<<<256, 256, 0, stream>>>(scores, attn);

  // 5) context = attn @ enc.
  context_kernel<<<256, 256, 0, stream>>>(enc, attn, ctx);

  // 6) Xg = [emb | ctx | h0]; gates = Xg @ Wcat^T + (b_ih+b_hh).
  build_xg_kernel<<<1536, 256, 0, stream>>>(x, emb, ctx, h0, Xg);
  gemm_bf16_tile<0><<<dim3(4, 16, 1), 256, 0, stream>>>(
      Xg, 1536, 0, Wcat, 1536, bsum, 0, nullptr, gates, 2048, 0, 1536);

  // 7) LSTM pointwise -> h1, c1 (written straight into d_out).
  lstm_kernel<<<512, 256, 0, stream>>>(gates, c0, h1, c1);

  // 8) Xf = [h1 | ctx | emb]; logits = Xf @ fc_W^T + fc_b.
  build_xf_kernel<<<1536, 256, 0, stream>>>(x, emb, ctx, h1, Xf);
  gemm_bf16_tile<0><<<dim3(4, 250, 1), 256, 0, stream>>>(
      Xf, 1536, 0, fc_W, 1536, fc_b, 0, nullptr, logits, 32000, 0, 1536);
}
